// ReasonDecoder_1236950582034
// MI455X (gfx1250) — matrix operand order repaired and checked by reading, not compiled, not run on hardware
//
#include <hip/hip_runtime.h>
#include <hip/hip_bf16.h>
#include <math.h>

// ---------------------------------------------------------------------------
// Types for CDNA5 WMMA (wave32)
// ---------------------------------------------------------------------------
typedef __attribute__((ext_vector_type(16))) __bf16 v16bf;
typedef __attribute__((ext_vector_type(8)))  __bf16 v8bf;
typedef __attribute__((ext_vector_type(8)))  float  v8f;
typedef __attribute__((ext_vector_type(4)))  unsigned int v4u;
typedef __attribute__((ext_vector_type(8)))  int v8i_t;
typedef __attribute__((ext_vector_type(4)))  int v4i_t;

union FragBF {            // assemble a 16-elem WMMA fragment from 2 x b128
    v16bf v;
    v8bf  h[2];
};

#define TM 128            // M tile (output pixels) per block
#define TN 64             // N tile (output channels) per block
#define TK 32             // K step

// ---------------------------------------------------------------------------
// f32 -> bf16 conversion (weights / activations)
// ---------------------------------------------------------------------------
__global__ void cvt_f32_bf16_kernel(const float* __restrict__ in,
                                    __bf16* __restrict__ out, int n) {
    int i = blockIdx.x * 256 + threadIdx.x;
    if (i < n) out[i] = (__bf16)in[i];
}

// ---------------------------------------------------------------------------
// x0 = image_embeddings + pos2d(C,H,W), stored bf16 NCHW [8][512][1024]
// ---------------------------------------------------------------------------
__global__ void build_x0_kernel(const float* __restrict__ img,
                                __bf16* __restrict__ x0) {
    int idx = blockIdx.x * 256 + threadIdx.x;           // 8*512*1024
    int n = idx & 1023;
    int c = (idx >> 10) & 511;
    int h = n >> 5, w = n & 31;
    const float k = logf(10000.f) / 256.f;
    float pe;
    if (c < 256) {
        int i = c >> 1;
        float dv = expf(-(float)(2 * i) * k);
        float arg = (float)w * dv;
        pe = (c & 1) ? cosf(arg) : sinf(arg);
    } else {
        int cc = c - 256;
        int i = cc >> 1;
        float dv = expf(-(float)(2 * i) * k);
        float arg = (float)h * dv;
        pe = (cc & 1) ? cosf(arg) : sinf(arg);
    }
    x0[idx] = (__bf16)(img[idx] + pe);
}

// ---------------------------------------------------------------------------
// Implicit-GEMM conv (ks = 1 or 3, pad = ks/2) using v_wmma_f32_16x16x32_bf16
//   in  : bf16 NCHW [B][Cin][H*W]
//   wt  : bf16 [Cout][Cin*ks*ks]      (OIHW flattened == [n][k])
//   out : f32  NCHW [B][Cout][H*W]
// Block = 256 threads (8 waves). Tile: 128(M) x 64(N), K step 32.
// Weights are staged into LDS by the Tensor Data Mover (wave 0 issues the
// descriptor, TENSORcnt tracks completion); activations are vector-gathered
// (3x3 halo) in parallel. Grid: (HW/128, Cout/64, B).
// ---------------------------------------------------------------------------
__global__ __launch_bounds__(256)
void conv_wmma_kernel(const __bf16* __restrict__ in,
                      const __bf16* __restrict__ wt,
                      const float*  __restrict__ bias,
                      float* __restrict__ out,
                      int Cin, int Cout, int H, int W, int ks) {
    __shared__ __align__(16) __bf16 As[TM][TK];
    __shared__ __align__(16) __bf16 Bs[TN][TK];

    const int HW = H * W;
    const int K  = Cin * ks * ks;
    const int b  = blockIdx.z;
    const int m0 = blockIdx.x * TM;
    const int n0 = blockIdx.y * TN;
    const int tid  = threadIdx.x;
    const int wid  = tid >> 5;
    const int lane = tid & 31;
    const int half = lane >> 4;
    const int r    = lane & 15;

    v8f z = {0.f, 0.f, 0.f, 0.f, 0.f, 0.f, 0.f, 0.f};
    v8f acc[4];
    acc[0] = z; acc[1] = z; acc[2] = z; acc[3] = z;

    const __bf16* inb = in + (size_t)b * Cin * HW;

#if __has_builtin(__builtin_amdgcn_tensor_load_to_lds) && \
    __has_builtin(__builtin_amdgcn_s_wait_tensorcnt)
    // --- Tensor DMA Descriptor (D#) for the 64x32 bf16 weight tile ---------
    const unsigned lds_b = (unsigned)(size_t)(void*)&Bs[0][0];
    const unsigned long long gbase =
        (unsigned long long)(size_t)(wt + (size_t)n0 * K);
    v8i_t g1;
    g1[0] = (int)(1u << 16);                      // data_size=1 (2 bytes)
    g1[1] = (int)((unsigned)(K & 0xFFFF) << 16);  // tensor_dim0[15:0]
    g1[2] = (int)((unsigned)(K >> 16) | ((unsigned)(Cout & 0xFFFF) << 16));
    g1[3] = (int)((unsigned)(Cout >> 16) | ((unsigned)TK << 16)); // tile_dim0
    g1[4] = (int)TN;                              // tile_dim1=64, tile_dim2=0
    g1[5] = (int)K;                               // tensor_dim0_stride lo32
    g1[6] = 0;                                    // stride hi16 / dim1_stride
    g1[7] = 0;
    v4i_t g2 = {0, 0, 0, 0};
    v4i_t g3 = {0, 0, 0, 0};
    v8i_t g4 = {0, 0, 0, 0, 0, 0, 0, 0};
    const bool use_tdm = true;
#else
    const bool use_tdm = false;
#endif

    for (int k0 = 0; k0 < K; k0 += TK) {
#if __has_builtin(__builtin_amdgcn_tensor_load_to_lds) && \
    __has_builtin(__builtin_amdgcn_s_wait_tensorcnt)
        if (wid == 0) {                           // one wave drives the TDM
            unsigned long long ga = gbase + (unsigned long long)k0 * 2ull;
            v4u g0;
            g0[0] = 1u;                           // count=1 (valid descriptor)
            g0[1] = lds_b;                        // LDS destination offset
            g0[2] = (unsigned)(ga & 0xFFFFFFFFu); // global_addr[31:0]
            g0[3] = (unsigned)((ga >> 32) & 0x1FFFFFFu) | (2u << 30); // type=2
            __builtin_amdgcn_tensor_load_to_lds(g0, g1, g2, g3, g4, 0);
        }
#endif
        // ---- vector-gather the 128x32 activation tile while TDM runs ------
        #pragma unroll
        for (int i = 0; i < 16; i++) {
            int idx = i * 256 + tid;              // 0..4095
            int ml  = idx >> 5;                   // row in tile
            int kl  = idx & 31;                   // k within step
            int kg  = k0 + kl;
            __bf16 v = (__bf16)0.0f;
            int m = m0 + ml;
            if (ks == 1) {
                v = inb[(size_t)kg * HW + m];
            } else {
                int c  = kg / 9;
                int rr = kg - c * 9;
                int kh = rr / 3, kw = rr - kh * 3;
                int hh = m / W + kh - 1;
                int ww = m - (m / W) * W + kw - 1;
                if (hh >= 0 && hh < H && ww >= 0 && ww < W)
                    v = inb[((size_t)c * H + hh) * W + ww];
            }
            As[ml][kl] = v;
        }
        if (!use_tdm) {
            // fallback: vector-load the weight tile (2 x b128 per thread)
            int row = tid >> 1, hh = tid & 1;
            if (row < TN) {
                v8bf* dst = (v8bf*)&Bs[row][hh * 16];
                const v8bf* srcp =
                    (const v8bf*)(wt + (size_t)(n0 + row) * K + k0 + hh * 16);
                dst[0] = srcp[0];
                dst[1] = srcp[1];
            }
        }
#if __has_builtin(__builtin_amdgcn_tensor_load_to_lds) && \
    __has_builtin(__builtin_amdgcn_s_wait_tensorcnt)
        if (wid == 0) __builtin_amdgcn_s_wait_tensorcnt(0);
#endif
        __syncthreads();

        // ---- A fragment: lane<16 holds K{0..7,16..23}, lane>=16 K{8..15,24..31}
        FragBF af;
        int arow = wid * 16 + r;
        int ko = half ? 8 : 0;
        af.h[0] = *(const v8bf*)&As[arow][ko];        // ds_load_b128
        af.h[1] = *(const v8bf*)&As[arow][16 + ko];   // ds_load_b128

        // ---- 4 N-subtiles per wave ----
        #pragma unroll
        for (int t = 0; t < 4; t++) {
            FragBF bf;
            int bcol = t * 16 + r;
            bf.h[0] = *(const v8bf*)&Bs[bcol][half * 8];
            bf.h[1] = *(const v8bf*)&Bs[bcol][16 + half * 8];
            acc[t] = __builtin_amdgcn_wmma_f32_16x16x32_bf16(
                false, af.v, false, bf.v, (short)0, acc[t], false, false);
        }
        __syncthreads();
    }

    // ---- store D: VGPR e -> M = e (+8 for lanes 16..31), N = lane&15 ----
    #pragma unroll
    for (int t = 0; t < 4; t++) {
        int n = n0 + t * 16 + r;
        int mbase = m0 + wid * 16 + (half ? 8 : 0);
        float* op = out + ((size_t)b * Cout + n) * HW + mbase;
        float bv = bias ? bias[n] : 0.f;
        float4 lo = make_float4(acc[t][0] + bv, acc[t][1] + bv,
                                acc[t][2] + bv, acc[t][3] + bv);
        float4 hi = make_float4(acc[t][4] + bv, acc[t][5] + bv,
                                acc[t][6] + bv, acc[t][7] + bv);
        *(float4*)(op)     = lo;
        *(float4*)(op + 4) = hi;
    }
}

// ---------------------------------------------------------------------------
// BatchNorm statistics: per-channel mean / rsqrt(var+eps) over (B,H,W)
// ---------------------------------------------------------------------------
__global__ __launch_bounds__(256)
void bn_stats_kernel(const float* __restrict__ x, float* __restrict__ mean,
                     float* __restrict__ rstd, int C, int HW, int B) {
    int c = blockIdx.x;
    __shared__ float s1[256], s2[256];
    float a = 0.f, q = 0.f;
    int N = B * HW;
    for (int i = threadIdx.x; i < N; i += 256) {
        int bb = i / HW, p = i - bb * HW;
        float v = x[((size_t)bb * C + c) * HW + p];
        a += v; q += v * v;
    }
    s1[threadIdx.x] = a; s2[threadIdx.x] = q;
    __syncthreads();
    for (int s = 128; s > 0; s >>= 1) {
        if (threadIdx.x < s) {
            s1[threadIdx.x] += s1[threadIdx.x + s];
            s2[threadIdx.x] += s2[threadIdx.x + s];
        }
        __syncthreads();
    }
    if (threadIdx.x == 0) {
        float m = s1[0] / N;
        float v = s2[0] / N - m * m;
        mean[c] = m;
        rstd[c] = rsqrtf(v + 1e-5f);
    }
}

__global__ void bn_relu_f32_kernel(float* __restrict__ x,
                                   const float* __restrict__ mean,
                                   const float* __restrict__ rstd,
                                   const float* __restrict__ g,
                                   const float* __restrict__ bb,
                                   int C, int HW, int total) {
    int idx = blockIdx.x * 256 + threadIdx.x;
    if (idx >= total) return;
    int c = (idx / HW) % C;
    float y = (x[idx] - mean[c]) * rstd[c] * g[c] + bb[c];
    x[idx] = fmaxf(y, 0.f);
}

__global__ void bn_relu_bf16_kernel(const float* __restrict__ x,
                                    __bf16* __restrict__ y,
                                    const float* __restrict__ mean,
                                    const float* __restrict__ rstd,
                                    const float* __restrict__ g,
                                    const float* __restrict__ bb,
                                    int C, int HW, int total) {
    int idx = blockIdx.x * 256 + threadIdx.x;
    if (idx >= total) return;
    int c = (idx / HW) % C;
    float v = (x[idx] - mean[c]) * rstd[c] * g[c] + bb[c];
    y[idx] = (__bf16)fmaxf(v, 0.f);
}

// ---------------------------------------------------------------------------
// Bilinear 2x upsample (half-pixel, matches jax.image.resize), f32 -> bf16
// ---------------------------------------------------------------------------
__global__ void upsample2x_kernel(const float* __restrict__ in,
                                  __bf16* __restrict__ out,
                                  int C, int H, int W) {
    int W2 = 2 * W, H2 = 2 * H;
    size_t idx = (size_t)blockIdx.x * 256 + threadIdx.x;
    int wo = (int)(idx % W2);
    size_t rr = idx / W2;
    int ho = (int)(rr % H2);
    size_t bc = rr / H2;                      // b*C + c
    const float* ip = in + bc * (size_t)(H * W);
    float cy = 0.5f * ho - 0.25f;
    float cx = 0.5f * wo - 0.25f;
    int h0 = (int)floorf(cy); float fy = cy - h0;
    int w0 = (int)floorf(cx); float fx = cx - w0;
    int h1 = h0 + 1 < H ? h0 + 1 : H - 1; if (h0 < 0) h0 = 0;
    int w1 = w0 + 1 < W ? w0 + 1 : W - 1; if (w0 < 0) w0 = 0;
    float v00 = ip[h0 * W + w0], v01 = ip[h0 * W + w1];
    float v10 = ip[h1 * W + w0], v11 = ip[h1 * W + w1];
    float v = (1.f - fy) * ((1.f - fx) * v00 + fx * v01) +
              fy * ((1.f - fx) * v10 + fx * v11);
    out[idx] = (__bf16)v;
}

// ---------------------------------------------------------------------------
// ans = mean over H*W of src; thread per (b,c)
// ---------------------------------------------------------------------------
__global__ void ans_kernel(const float* __restrict__ src,
                           float* __restrict__ ans) {
    int idx = blockIdx.x * 256 + threadIdx.x;   // 4096
    if (idx >= 4096) return;
    const float* p = src + (size_t)idx * 1024;
    float s = 0.f;
    for (int i = 0; i < 1024; i++) s += p[i];
    ans[idx] = s * (1.f / 1024.f);
}

// ans_vec = ans @ ln_ans_w.T + ln_ans_b
__global__ void ansvec_kernel(const float* __restrict__ ans,
                              const float* __restrict__ w,
                              const float* __restrict__ bias,
                              float* __restrict__ out) {
    int idx = blockIdx.x * 256 + threadIdx.x;   // 8*512
    if (idx >= 4096) return;
    int b = idx >> 9, d = idx & 511;
    float s = bias[d];
    const float* wr = w + (size_t)d * 512;
    const float* ar = ans + b * 512;
    for (int k = 0; k < 512; k++) s += ar[k] * wr[k];
    out[idx] = s;
}

// ---------------------------------------------------------------------------
// qa = [state, ans_vec]; o = softmax(qa)*qa; word = o[:512] + o[512:]
// ---------------------------------------------------------------------------
__global__ __launch_bounds__(256)
void qa_word_kernel(const float* __restrict__ state,
                    const float* __restrict__ ansvec,
                    float* __restrict__ word) {
    int b = blockIdx.x, t = threadIdx.x;
    __shared__ float qa[1024];
    __shared__ float red[256];
    for (int i = t; i < 1024; i += 256)
        qa[i] = (i < 512) ? state[b * 512 + i] : ansvec[b * 512 + i - 512];
    __syncthreads();
    float mx = -1e30f;
    for (int i = t; i < 1024; i += 256) mx = fmaxf(mx, qa[i]);
    red[t] = mx; __syncthreads();
    for (int s = 128; s > 0; s >>= 1) {
        if (t < s) red[t] = fmaxf(red[t], red[t + s]);
        __syncthreads();
    }
    mx = red[0]; __syncthreads();
    float sm = 0.f;
    for (int i = t; i < 1024; i += 256) sm += expf(qa[i] - mx);
    red[t] = sm; __syncthreads();
    for (int s = 128; s > 0; s >>= 1) {
        if (t < s) red[t] += red[t + s];
        __syncthreads();
    }
    float den = red[0];
    for (int j = t; j < 512; j += 256) {
        float o1 = expf(qa[j] - mx) / den * qa[j];
        float o2 = expf(qa[j + 512] - mx) / den * qa[j + 512];
        word[b * 512 + j] = o1 + o2;
    }
}

// txtout = word @ txt_w.T + txt_b   -> (8, 2305): dyn conv weights + bias
__global__ void txt_kernel(const float* __restrict__ word,
                           const float* __restrict__ tw,
                           const float* __restrict__ tb,
                           float* __restrict__ out) {
    int idx = blockIdx.x * 256 + threadIdx.x;   // 8*2305
    if (idx >= 8 * 2305) return;
    int b = idx / 2305, o = idx - b * 2305;
    float s = tb[o];
    const float* wr = tw + (size_t)o * 512;
    const float* wb = word + b * 512;
    for (int k = 0; k < 512; k++) s += wb[k] * wr[k];
    out[idx] = s;
}

// ---------------------------------------------------------------------------
// Per-batch dynamic 3x3 conv (groups=B, 256 in-ch -> 1 out-ch) -> masks
// ---------------------------------------------------------------------------
__global__ __launch_bounds__(256)
void dyn_conv_kernel(const float* __restrict__ x3,
                     const float* __restrict__ txtout,
                     float* __restrict__ masks) {
    int b = blockIdx.y;
    int p = blockIdx.x * 256 + threadIdx.x;     // < 16384
    int ho = p >> 7, wo = p & 127;
    const float* wv = txtout + b * 2305;
    float acc = wv[2304];
    const float* xb = x3 + (size_t)b * 256 * 16384;
    for (int c = 0; c < 256; c++) {
        const float* xc = xb + (size_t)c * 16384;
        const float* wc = wv + c * 9;
        #pragma unroll
        for (int kh = 0; kh < 3; kh++) {
            int h = ho + kh - 1;
            if (h < 0 || h >= 128) continue;
            #pragma unroll
            for (int kw = 0; kw < 3; kw++) {
                int w = wo + kw - 1;
                if (w < 0 || w >= 128) continue;
                acc += xc[h * 128 + w] * wc[kh * 3 + kw];
            }
        }
    }
    masks[(size_t)b * 16384 + p] = acc;
}

// ---------------------------------------------------------------------------
// Router + experts -> logits (8, 23); one block per batch
// ---------------------------------------------------------------------------
__global__ __launch_bounds__(256)
void router_kernel(const float* __restrict__ state,
                   const float* __restrict__ ansvec,
                   const float* __restrict__ rw1, const float* __restrict__ rb1,
                   const float* __restrict__ rw2, const float* __restrict__ rb2,
                   const float* __restrict__ lw2, const float* __restrict__ lb2,
                   const float* __restrict__ ew,  const float* __restrict__ eb,
                   float* __restrict__ logits) {
    int b = blockIdx.x, t = threadIdx.x;
    __shared__ float qa[1024];
    __shared__ float hid[256];
    __shared__ float gl[8];
    __shared__ float feat[128];
    for (int i = t; i < 1024; i += 256)
        qa[i] = (i < 512) ? state[b * 512 + i] : ansvec[b * 512 + i - 512];
    __syncthreads();
    {
        float s = rb1[t];
        const float* wr = rw1 + (size_t)t * 1024;
        for (int k = 0; k < 1024; k++) s += qa[k] * wr[k];
        hid[t] = fmaxf(s, 0.f);
    }
    __syncthreads();
    if (t < 8) {
        float s = rb2[t];
        const float* wr = rw2 + t * 256;
        for (int k = 0; k < 256; k++) s += hid[k] * wr[k];
        gl[t] = s;
    }
    __syncthreads();
    if (t == 0) {
        float mx = gl[0];
        for (int e = 1; e < 8; e++) mx = fmaxf(mx, gl[e]);
        float den = 0.f;
        for (int e = 0; e < 8; e++) { gl[e] = expf(gl[e] - mx); den += gl[e]; }
        for (int e = 0; e < 8; e++) gl[e] /= den;
    }
    if (t < 128) {
        float s = lb2[t];
        const float* wr = lw2 + (size_t)t * 512;
        for (int k = 0; k < 512; k++) s += ansvec[b * 512 + k] * wr[k];
        feat[t] = s;
    }
    __syncthreads();
    if (t < 23) {
        float s = 0.f;
        for (int e = 0; e < 8; e++) {
            float d = eb[e * 23 + t];
            const float* we = ew + (size_t)(e * 23 + t) * 128;
            for (int f = 0; f < 128; f++) d += we[f] * feat[f];
            s += gl[e] * d;
        }
        logits[b * 23 + t] = s;
    }
}

// ---------------------------------------------------------------------------
// Host orchestration
// ---------------------------------------------------------------------------
extern "C" void kernel_launch(void* const* d_in, const int* in_sizes, int n_in,
                              void* d_out, int out_size, void* d_ws, size_t ws_size,
                              hipStream_t stream) {
    const float* img   = (const float*)d_in[0];
    const float* state = (const float*)d_in[2];
    const float* W_t   = (const float*)d_in[4];
    const float* w1    = (const float*)d_in[7];
    const float* g1    = (const float*)d_in[8];
    const float* b1    = (const float*)d_in[9];
    const float* w2    = (const float*)d_in[10];
    const float* g2    = (const float*)d_in[11];
    const float* b2    = (const float*)d_in[12];
    const float* w3    = (const float*)d_in[13];
    const float* b3    = (const float*)d_in[14];
    const float* txtw  = (const float*)d_in[15];
    const float* txtb  = (const float*)d_in[16];
    const float* lnw   = (const float*)d_in[17];
    const float* lnb   = (const float*)d_in[18];
    const float* ln2w  = (const float*)d_in[19];
    const float* ln2b  = (const float*)d_in[20];
    const float* rw1   = (const float*)d_in[21];
    const float* rb1   = (const float*)d_in[22];
    const float* rw2   = (const float*)d_in[23];
    const float* rb2   = (const float*)d_in[24];
    const float* ew    = (const float*)d_in[25];
    const float* eb    = (const float*)d_in[26];

    float* out    = (float*)d_out;
    float* masks  = out;                        // (8,1,128,128) = 131072
    float* logits = out + 131072;               // (8,23) = 184
    float* src    = out + 131072 + 184;         // (8,512,32,32) = 4194304

    char* ws = (char*)d_ws;
    size_t off = 0;
    auto alloc = [&](size_t bytes) -> char* {
        char* p = ws + off;
        off += (bytes + 255) & ~(size_t)255;
        return p;
    };
    __bf16* wtbf  = (__bf16*)alloc((size_t)512 * 512 * 2);
    __bf16* w1bf  = (__bf16*)alloc((size_t)512 * 4608 * 2);
    __bf16* w2bf  = (__bf16*)alloc((size_t)256 * 4608 * 2);
    __bf16* w3bf  = (__bf16*)alloc((size_t)256 * 256 * 2);
    float*  mean1 = (float*)alloc(512 * 4);
    float*  rstd1 = (float*)alloc(512 * 4);
    float*  mean2 = (float*)alloc(256 * 4);
    float*  rstd2 = (float*)alloc(256 * 4);
    float*  ansb  = (float*)alloc(4096 * 4);
    float*  ansv  = (float*)alloc(4096 * 4);
    float*  word  = (float*)alloc(8 * 512 * 4);
    float*  txto  = (float*)alloc(8 * 2305 * 4);
    char*   MED   = alloc(67108864);    // x0 bf16 (8.4MB) -> bn2 bf16 (67MB)
    char*   BIG1  = alloc(134217728);   // up1 bf16 -> up2 bf16 -> x3 f32
    char*   BIG2  = alloc(134217728);   // conv1 f32 -> conv2 f32

    // 1. weights -> bf16
    cvt_f32_bf16_kernel<<<(512 * 512 + 255) / 256, 256, 0, stream>>>(W_t, wtbf, 512 * 512);
    cvt_f32_bf16_kernel<<<(512 * 4608 + 255) / 256, 256, 0, stream>>>(w1, w1bf, 512 * 4608);
    cvt_f32_bf16_kernel<<<(256 * 4608 + 255) / 256, 256, 0, stream>>>(w2, w2bf, 256 * 4608);
    cvt_f32_bf16_kernel<<<(256 * 256 + 255) / 256, 256, 0, stream>>>(w3, w3bf, 256 * 256);

    // 2. x0 = img + pos2d (bf16 NCHW)
    build_x0_kernel<<<(8 * 512 * 1024) / 256, 256, 0, stream>>>(img, (__bf16*)MED);

    // 3. src = x0 @ W_t^T  (WMMA, 1x1)  -> d_out src slot
    conv_wmma_kernel<<<dim3(8, 8, 8), 256, 0, stream>>>(
        (__bf16*)MED, wtbf, nullptr, src, 512, 512, 32, 32, 1);

    // 4-5. ans / ans_vec
    ans_kernel<<<16, 256, 0, stream>>>(src, ansb);
    ansvec_kernel<<<16, 256, 0, stream>>>(ansb, lnw, lnb, ansv);

    // 6. up2x(src) -> bf16 @64x64
    upsample2x_kernel<<<(8 * 512 * 4096) / 256, 256, 0, stream>>>(
        src, (__bf16*)BIG1, 512, 32, 32);

    // 7. conv1 3x3 512->512 (WMMA + TDM weights)
    conv_wmma_kernel<<<dim3(32, 8, 8), 256, 0, stream>>>(
        (__bf16*)BIG1, w1bf, nullptr, (float*)BIG2, 512, 512, 64, 64, 3);

    // 8-9. BN1 + ReLU (in place, f32)
    bn_stats_kernel<<<512, 256, 0, stream>>>((float*)BIG2, mean1, rstd1, 512, 4096, 8);
    bn_relu_f32_kernel<<<(8 * 512 * 4096) / 256, 256, 0, stream>>>(
        (float*)BIG2, mean1, rstd1, g1, b1, 512, 4096, 8 * 512 * 4096);

    // 10. up2x -> bf16 @128x128
    upsample2x_kernel<<<(8 * 512 * 16384) / 256, 256, 0, stream>>>(
        (float*)BIG2, (__bf16*)BIG1, 512, 64, 64);

    // 11. conv2 3x3 512->256 (WMMA + TDM weights) — dominant compute
    conv_wmma_kernel<<<dim3(128, 4, 8), 256, 0, stream>>>(
        (__bf16*)BIG1, w2bf, nullptr, (float*)BIG2, 512, 256, 128, 128, 3);

    // 12-13. BN2 + ReLU -> bf16
    bn_stats_kernel<<<256, 256, 0, stream>>>((float*)BIG2, mean2, rstd2, 256, 16384, 8);
    bn_relu_bf16_kernel<<<(8 * 256 * 16384) / 256, 256, 0, stream>>>(
        (float*)BIG2, (__bf16*)MED, mean2, rstd2, g2, b2, 256, 16384, 8 * 256 * 16384);

    // 14. conv3 1x1 256->256 + bias (WMMA) -> x3 f32
    conv_wmma_kernel<<<dim3(128, 4, 8), 256, 0, stream>>>(
        (__bf16*)MED, w3bf, b3, (float*)BIG1, 256, 256, 128, 128, 1);

    // 15-16. word -> dynamic conv params
    qa_word_kernel<<<8, 256, 0, stream>>>(state, ansv, word);
    txt_kernel<<<(8 * 2305 + 255) / 256, 256, 0, stream>>>(word, txtw, txtb, txto);

    // 17. dynamic grouped conv -> masks
    dyn_conv_kernel<<<dim3(64, 8), 256, 0, stream>>>((float*)BIG1, txto, masks);

    // 18. router + experts -> logits
    router_kernel<<<8, 256, 0, stream>>>(state, ansv, rw1, rb1, rw2, rb2,
                                         ln2w, ln2b, ew, eb, logits);
}
